// Decoder_10170482557258
// MI455X (gfx1250) — matrix operand order repaired and checked by reading, not compiled, not run on hardware
//
#include <hip/hip_runtime.h>
#include <hip/hip_bf16.h>

// Problem constants (match reference)
#define TT  32
#define BB  128
#define VV  32000
#define EE  256
#define HH  512
#define DWW 300
#define GG  (4*HH)   // 2048

// LDS B-panel: 64 rows x 32 K bf16 (64B/row), padded to 80B/row for bank spread.
#define LDS_ROW 80
#define LDS_PANEL (64 * LDS_ROW)

#if defined(__has_builtin) && __has_builtin(__builtin_amdgcn_global_load_async_to_lds_b128)
#define HAVE_ASYNC_LDS 1
#else
#define HAVE_ASYNC_LDS 0
#endif

typedef __attribute__((ext_vector_type(16))) __bf16 v16bf;
typedef __attribute__((ext_vector_type(8)))  float  v8f;
typedef __attribute__((ext_vector_type(4)))  float  v4f;   // native vector for NT ld/st
typedef int v4i __attribute__((vector_size(16)));   // matches builtin's V4i param

union Frag { v16bf v; uint4 q[2]; };

__device__ __forceinline__ float sigm(float x) { return 1.0f / (1.0f + __expf(-x)); }

__device__ __forceinline__ void wait_async_le2() {
#if HAVE_ASYNC_LDS
#if defined(__has_builtin) && __has_builtin(__builtin_amdgcn_s_wait_asynccnt)
  __builtin_amdgcn_s_wait_asynccnt(2);
#else
  asm volatile("s_wait_asynccnt 0x2" ::: "memory");
#endif
#endif
}
__device__ __forceinline__ void wait_async_le0() {
#if HAVE_ASYNC_LDS
#if defined(__has_builtin) && __has_builtin(__builtin_amdgcn_s_wait_asynccnt)
  __builtin_amdgcn_s_wait_asynccnt(0);
#else
  asm volatile("s_wait_asynccnt 0x0" ::: "memory");
#endif
#endif
}

// Stage one 64x32 bf16 B panel (K offset kb) into LDS. Thread tid copies 32B:
// row = tid>>1 (0..63), chunk = tid&1 covering K range [chunk*16, chunk*16+16).
__device__ __forceinline__ void stage_b_panel(const __hip_bfloat16* __restrict__ Bm,
                                              size_t rowBase, int K, int kb,
                                              char* lbuf, int tid) {
  const int srow = tid >> 1, schunk = tid & 1;
  const char* src = (const char*)(Bm + rowBase + (size_t)srow * K + kb + schunk * 16);
  char* dst = lbuf + srow * LDS_ROW + schunk * 32;
#if HAVE_ASYNC_LDS
  auto gsrc = (__attribute__((address_space(1))) v4i*)(uintptr_t)src;
  auto ldst = (__attribute__((address_space(3))) v4i*)(unsigned)(uintptr_t)dst;
  __builtin_amdgcn_global_load_async_to_lds_b128(gsrc,     ldst,     0, 0);
  __builtin_amdgcn_global_load_async_to_lds_b128(gsrc + 1, ldst + 1, 0, 0);
#else
  uint4 t0 = *(const uint4*)src;
  uint4 t1 = *(const uint4*)(src + 16);
  *(uint4*)dst = t0;
  *(uint4*)(dst + 16) = t1;
#endif
}

// Load all 4 B fragments first (in-order DS retirement -> partial dscnt waits),
// then issue the 4 WMMAs so fragment fetch overlaps WMMA issue.
__device__ __forceinline__ void wmma4(const Frag& a, const char* lbuf,
                                      v8f acc[4], int nl, int half) {
  Frag b[4];
#pragma unroll
  for (int j = 0; j < 4; ++j) {
    const char* bp = lbuf + (j * 16 + nl) * LDS_ROW + half * 16;
    b[j].q[0] = *(const uint4*)bp;
    b[j].q[1] = *(const uint4*)(bp + 32);
  }
#pragma unroll
  for (int j = 0; j < 4; ++j) {
    acc[j] = __builtin_amdgcn_wmma_f32_16x16x32_bf16(
        false, a.v, false, b[j].v, (short)0, acc[j], false, false);
  }
}

// ---------------- conversion / setup kernels ----------------

__global__ void f32_to_bf16_kernel(const float* __restrict__ src,
                                   __hip_bfloat16* __restrict__ dst, int n) {
  int i = blockIdx.x * blockDim.x + threadIdx.x;
  int stride = gridDim.x * blockDim.x;
  for (; i < n; i += stride) dst[i] = __float2bfloat16(src[i]);
}

__global__ void bias_comb_kernel(const float* __restrict__ a,
                                 const float* __restrict__ b,
                                 float* __restrict__ o, int n) {
  int i = blockIdx.x * blockDim.x + threadIdx.x;
  if (i < n) o[i] = a[i] + b[i];
}

__global__ void gather_emb_kernel(const int* __restrict__ x_seq,
                                  const float* __restrict__ embedding,
                                  __hip_bfloat16* __restrict__ ax) {
  int row = blockIdx.x;
  int k   = threadIdx.x;
  int idx = x_seq[row];
  ax[(size_t)row * EE + k] = __float2bfloat16(embedding[(size_t)idx * EE + k]);
}

__global__ void init_h0c0_kernel(const float* __restrict__ emb,
                                 const int* __restrict__ projectEmb,
                                 const float* __restrict__ W_h, const float* __restrict__ b_h,
                                 const float* __restrict__ W_c, const float* __restrict__ b_c,
                                 float* __restrict__ h0, float* __restrict__ c0,
                                 __hip_bfloat16* __restrict__ h_bf) {
  int i = blockIdx.x * blockDim.x + threadIdx.x;   // b*H + j
  int b = i / HH, j = i % HH;
  float h, c;
  if (*projectEmb) {
    float ah = b_h[j], ac = b_c[j];
    const float* e  = emb + (size_t)b * DWW;
    const float* wh = W_h + (size_t)j * DWW;
    const float* wc = W_c + (size_t)j * DWW;
    for (int k = 0; k < DWW; ++k) { float ev = e[k]; ah += ev * wh[k]; ac += ev * wc[k]; }
    h = ah; c = ac;
  } else {
    h = emb[i]; c = emb[(size_t)BB * HH + i];
  }
  h0[i] = h; c0[i] = c;
  h_bf[i] = __float2bfloat16(h);
}

// ---------------- WMMA bf16 GEMM (LDS-staged, double-buffered) ----------------
// C[M,N](f32,ldc) = A[M,K](bf16,rm) * Bm[N,K](bf16,rm)^T (+bias[N]) (+addend[M,ldadd])
// Block: 128 thr = 4 wave32; wave w: rows [by*64+w*16,+16), cols [bx*64,+64).
// B panels staged to LDS (async copy when available), A double-buffered in regs.
// ntStore=1: stream C with non-temporal stores (don't pollute L2).
__global__ __launch_bounds__(128)
void gemm_bf16_wmma(const __hip_bfloat16* __restrict__ A,
                    const __hip_bfloat16* __restrict__ Bm,
                    float* __restrict__ C,
                    const float* __restrict__ bias,
                    const float* __restrict__ addend,
                    int K, int ldc, int ldadd, int ntStore) {
  __shared__ __align__(16) char lds0[LDS_PANEL];
  __shared__ __align__(16) char lds1[LDS_PANEL];

  const int tid  = threadIdx.x;
  const int lane = tid & 31;
  const int wave = tid >> 5;
  const int m0 = blockIdx.y * 64 + wave * 16;
  const int n0 = blockIdx.x * 64;
  const int nl   = lane & 15;
  const int half = lane >> 4;
  const size_t rowBase = (size_t)n0 * K;

  // A fragment source: lane -> row m0+nl; VGPR0-3: K = half*8+0..7, VGPR4-7: +16.
  const __hip_bfloat16* aRow = A + (size_t)(m0 + nl) * K + half * 8;

  v8f acc[4];
#pragma unroll
  for (int j = 0; j < 4; ++j) acc[j] = (v8f){0.f,0.f,0.f,0.f,0.f,0.f,0.f,0.f};

  Frag a0, a1;
  // prologue: stage first panel, load first A frag
  stage_b_panel(Bm, rowBase, K, 0, lds0, tid);
  a0.q[0] = *(const uint4*)(aRow + 0);
  a0.q[1] = *(const uint4*)(aRow + 16);

  for (int kb = 0; kb < K; kb += 64) {
    __builtin_prefetch(aRow + kb + 128, 0, 3);
    // issue stage for the odd half into lds1; prefetch next A frag
    stage_b_panel(Bm, rowBase, K, kb + 32, lds1, tid);
    a1.q[0] = *(const uint4*)(aRow + kb + 32);
    a1.q[1] = *(const uint4*)(aRow + kb + 48);

    wait_async_le2();          // lds0's stage (prev iter / prologue) retired
    __syncthreads();           // lds0 visible to all waves
    wmma4(a0, lds0, acc, nl, half);
    __syncthreads();           // all waves done reading lds0

    const bool more = (kb + 64 < K);
    if (more) {
      stage_b_panel(Bm, rowBase, K, kb + 64, lds0, tid);
      a0.q[0] = *(const uint4*)(aRow + kb + 64);
      a0.q[1] = *(const uint4*)(aRow + kb + 80);
      wait_async_le2();        // lds1's stage retired (one newer batch in flight)
    } else {
      wait_async_le0();        // final: drain
    }
    __syncthreads();           // lds1 visible
    wmma4(a1, lds1, acc, nl, half);
    __syncthreads();           // all waves done reading lds1
  }

  // Store: C/D layout — lane L: col = L&15, VGPR i -> row = (L>>4)*8 + i.
#pragma unroll
  for (int j = 0; j < 4; ++j) {
    const int col = n0 + j * 16 + nl;
    const float badd = bias ? bias[col] : 0.0f;
#pragma unroll
    for (int i = 0; i < 8; ++i) {
      const int row = m0 + half * 8 + i;
      float v = acc[j][i] + badd;
      if (addend) v += addend[(size_t)row * ldadd + col];
      float* dst = &C[(size_t)row * ldc + col];
      if (ntStore) __builtin_nontemporal_store(v, dst);
      else         *dst = v;
    }
  }
}

// ---------------- LSTM pointwise ----------------
__global__ void lstm_pointwise_kernel(const float* __restrict__ gates,  // [B, 4H] i,f,g,o
                                      float* __restrict__ h, float* __restrict__ c,
                                      __hip_bfloat16* __restrict__ h_bf,
                                      __hip_bfloat16* __restrict__ h_all_t) {
  int i = blockIdx.x * blockDim.x + threadIdx.x;   // b*H + j
  int b = i / HH, j = i % HH;
  const float* g = gates + (size_t)b * GG;
  float gi = g[j], gf = g[HH + j], gg = g[2 * HH + j], go = g[3 * HH + j];
  float cn = sigm(gf) * c[i] + sigm(gi) * tanhf(gg);
  float hn = sigm(go) * tanhf(cn);
  c[i] = cn; h[i] = hn;
  __hip_bfloat16 hb = __float2bfloat16(hn);
  h_bf[i] = hb;
  h_all_t[i] = hb;
}

// ---------------- log-softmax ----------------
__global__ void row_lse_kernel(const float* __restrict__ logits, float* __restrict__ lse) {
  const int r = blockIdx.x;
  const float* x = logits + (size_t)r * VV;
  float m = -3.4e38f, s = 0.0f;
  for (int v = threadIdx.x; v < VV; v += blockDim.x) {
    float t = __builtin_nontemporal_load(&x[v]);   // streaming read, keep L2 for weights
    if (t > m) { s = s * __expf(m - t) + 1.0f; m = t; }
    else       { s += __expf(t - m); }
  }
  __shared__ float sm[256], ss[256];
  sm[threadIdx.x] = m; ss[threadIdx.x] = s;
  __syncthreads();
  for (int off = 128; off > 0; off >>= 1) {
    if ((int)threadIdx.x < off) {
      float m1 = sm[threadIdx.x], s1 = ss[threadIdx.x];
      float m2 = sm[threadIdx.x + off], s2 = ss[threadIdx.x + off];
      float M = fmaxf(m1, m2);
      ss[threadIdx.x] = s1 * __expf(m1 - M) + s2 * __expf(m2 - M);
      sm[threadIdx.x] = M;
    }
    __syncthreads();
  }
  if (threadIdx.x == 0) lse[r] = sm[0] + __logf(ss[0]);
}

__global__ void apply_lse_kernel(float* __restrict__ logits, const float* __restrict__ lse) {
  const int r = blockIdx.x;
  const float l = lse[r];
  v4f* p = (v4f*)(logits + (size_t)r * VV);
  for (int v = threadIdx.x; v < VV / 4; v += blockDim.x) {
    v4f t = __builtin_nontemporal_load(&p[v]);
    t = t - l;                      // scalar broadcast on native vector
    __builtin_nontemporal_store(t, &p[v]);
  }
}

// ---------------- launch ----------------
extern "C" void kernel_launch(void* const* d_in, const int* in_sizes, int n_in,
                              void* d_out, int out_size, void* d_ws, size_t ws_size,
                              hipStream_t stream) {
  const int*   x_seq      = (const int*)  d_in[0];
  const float* emb        = (const float*)d_in[1];
  const int*   projectEmb = (const int*)  d_in[2];
  const float* embedding  = (const float*)d_in[3];
  const float* W_ih       = (const float*)d_in[4];
  const float* b_ih       = (const float*)d_in[5];
  const float* W_hh       = (const float*)d_in[6];
  const float* b_hh       = (const float*)d_in[7];
  const float* W_out      = (const float*)d_in[8];
  const float* b_out      = (const float*)d_in[9];
  const float* W_h        = (const float*)d_in[10];
  const float* b_h        = (const float*)d_in[11];
  const float* W_c        = (const float*)d_in[12];
  const float* b_c        = (const float*)d_in[13];
  (void)in_sizes; (void)n_in; (void)out_size; (void)ws_size;

  char* ws = (char*)d_ws;
  size_t off = 0;
  auto alloc = [&](size_t bytes) -> void* {
    void* p = ws + off;
    off += (bytes + 255) & ~(size_t)255;
    return p;
  };
  __hip_bfloat16* Wout_bf = (__hip_bfloat16*)alloc((size_t)VV * HH * 2);
  __hip_bfloat16* Whh_bf  = (__hip_bfloat16*)alloc((size_t)GG * HH * 2);
  __hip_bfloat16* Wih_bf  = (__hip_bfloat16*)alloc((size_t)GG * EE * 2);
  __hip_bfloat16* Ax_bf   = (__hip_bfloat16*)alloc((size_t)TT * BB * EE * 2);
  __hip_bfloat16* hall_bf = (__hip_bfloat16*)alloc((size_t)TT * BB * HH * 2);
  __hip_bfloat16* hcur_bf = (__hip_bfloat16*)alloc((size_t)BB * HH * 2);
  float*          xg      = (float*)alloc((size_t)TT * BB * GG * 4);
  float*          gates   = (float*)alloc((size_t)BB * GG * 4);
  float*          b_comb  = (float*)alloc((size_t)GG * 4);
  float*          lse     = (float*)alloc((size_t)TT * BB * 4);

  float* logp  = (float*)d_out;                       // [T*B, V]
  float* h_out = logp + (size_t)TT * BB * VV;         // [B, H]
  float* c_out = h_out + (size_t)BB * HH;             // [B, H]

  auto cvt = [&](const float* s, __hip_bfloat16* d, int n) {
    int blocks = (n + 1023) / 1024; if (blocks > 8192) blocks = 8192;
    f32_to_bf16_kernel<<<blocks, 256, 0, stream>>>(s, d, n);
  };
  cvt(W_out, Wout_bf, VV * HH);
  cvt(W_hh,  Whh_bf,  GG * HH);
  cvt(W_ih,  Wih_bf,  GG * EE);

  bias_comb_kernel<<<GG / 256, 256, 0, stream>>>(b_ih, b_hh, b_comb, GG);

  gather_emb_kernel<<<TT * BB, EE, 0, stream>>>(x_seq, embedding, Ax_bf);

  init_h0c0_kernel<<<(BB * HH) / 256, 256, 0, stream>>>(
      emb, projectEmb, W_h, b_h, W_c, b_c, h_out, c_out, hcur_bf);

  // x_gates = Ax @ W_ih^T + (b_ih + b_hh)   [4096, 2048]
  gemm_bf16_wmma<<<dim3(GG / 64, (TT * BB) / 64), 128, 0, stream>>>(
      Ax_bf, Wih_bf, xg, b_comb, nullptr, EE, GG, 0, 0);

  // recurrence
  for (int t = 0; t < TT; ++t) {
    gemm_bf16_wmma<<<dim3(GG / 64, BB / 64), 128, 0, stream>>>(
        hcur_bf, Whh_bf, gates, nullptr, xg + (size_t)t * BB * GG, HH, GG, GG, 0);
    lstm_pointwise_kernel<<<(BB * HH) / 256, 256, 0, stream>>>(
        gates, h_out, c_out, hcur_bf, hall_bf + (size_t)t * BB * HH);
  }

  // logits = h_all @ W_out^T + b_out   [4096, 32000]  (streaming NT stores)
  gemm_bf16_wmma<<<dim3(VV / 64, (TT * BB) / 64), 128, 0, stream>>>(
      hall_bf, Wout_bf, logp, b_out, nullptr, HH, VV, 0, 1);

  // log_softmax over V
  row_lse_kernel<<<TT * BB, 256, 0, stream>>>(logp, lse);
  apply_lse_kernel<<<TT * BB, 256, 0, stream>>>(logp, lse);
}
